// GroupFFNExperts_18202071400827
// MI455X (gfx1250) — matrix operand order repaired and checked by reading, not compiled, not run on hardware
//
#include <hip/hip_runtime.h>
#include <math.h>

// Problem constants (from reference): E=64, CAP=1024, D=512, H=4*D=2048
#define E_   64
#define CAP_ 1024
#define D_   512
#define H_   2048

#define BR      128         // rows of X / Y per workgroup
#define HB      64          // H-chunk per iteration
#define THREADS 512         // 16 wave32 waves

// LDS pitches (elements), padded; pitch*2 bytes must be multiple of 16
#define XP  (D_ + 8)        // 520 -> 1040 B rows (16B aligned, 4-bank row skew)
#define W1P (D_ + 8)        // 520
#define TSP (HB + 8)        // 72  -> 144 B rows
#define W2P (HB + 8)        // 72

// LDS layout (bytes)
#define W1_BYTES (HB * W1P * 2)              // 66,560
#define TS_BYTES (BR * TSP * 2)              // 18,432
#define W2_BYTES (D_ * W2P * 2)              // 73,728
#define SMEM_BYTES (W1_BYTES + TS_BYTES + W2_BYTES)   // 158,720
// X staging (128 x 520 bf16 = 133,120 B) aliases the chunk buffers in phase 0.

typedef __bf16 v16bf __attribute__((ext_vector_type(16)));
typedef __bf16 v8bf  __attribute__((ext_vector_type(8)));
typedef float  v8f   __attribute__((ext_vector_type(8)));

// Build a 16-element bf16 WMMA fragment from two 16-byte LDS loads.
__device__ __forceinline__ v16bf frag2(const __bf16* p0, const __bf16* p1) {
  v8bf a = *(const v8bf*)p0;
  v8bf b = *(const v8bf*)p1;
  return __builtin_shufflevector(a, b, 0,1,2,3,4,5,6,7,8,9,10,11,12,13,14,15);
}

__device__ __forceinline__ unsigned pack2(float a, float b) {
  union { __bf16 h[2]; unsigned u; } x;
  x.h[0] = (__bf16)a; x.h[1] = (__bf16)b;
  return x.u;
}

__device__ __forceinline__ float gelu_exact(float x) {
  return 0.5f * x * (1.0f + erff(x * 0.70710678118654752f));
}

__global__ __launch_bounds__(THREADS)
void GroupFFNExperts_kernel(const float* __restrict__ xin,
                            const long long* __restrict__ valid_load,
                            const float* __restrict__ w1g,
                            const float* __restrict__ b1g,
                            const float* __restrict__ w2g,
                            const float* __restrict__ b2g,
                            float* __restrict__ out) {
  const int e        = blockIdx.y;
  const int rowStart = blockIdx.x * BR;
  const int tid      = threadIdx.x;
  const int valid    = (int)valid_load[e];

  float* outBlk = out + ((size_t)e * CAP_ + rowStart) * D_;

  // Fully-masked block: just write zeros (output buffer is poisoned).
  if (rowStart >= valid) {
    #pragma unroll 4
    for (int i = 0; i < (BR * D_ / 4) / THREADS; ++i) {
      int p = tid + i * THREADS;
      *(float4*)&outBlk[p * 4] = make_float4(0.f, 0.f, 0.f, 0.f);
    }
    return;
  }

  __shared__ __attribute__((aligned(16))) unsigned char smem[SMEM_BYTES];
  __bf16* const W1sT = (__bf16*)smem;                           // [HB][W1P]
  __bf16* const Ts   = (__bf16*)(smem + W1_BYTES);              // [BR][TSP]
  __bf16* const W2sT = (__bf16*)(smem + W1_BYTES + TS_BYTES);   // [D_][W2P]
  __bf16* const Xs   = (__bf16*)smem;                           // phase-0 alias

  const int lane = tid & 31;
  const int wv   = tid >> 5;       // wave id 0..15
  const int l16  = lane & 15;
  const int kg   = lane >> 4;      // K-half select per ISA fragment layout

  // GEMM1: T is BR x HB = 128x64 -> 8x4 tiles of 16x16; 2 column tiles/wave.
  const int tr  = wv & 7;          // T row tile (this wave's 16 X rows)
  const int tc0 = (wv >> 3) * 2;   // first of two T column tiles
  // GEMM2: wave grid 2x8 over Y(128x512); each wave owns a 64x64 tile.
  const int yr0 = (wv >> 3) * 64;
  const int yc0 = (wv & 7) * 64;

  // ---- Phase 0: stage X block (fp32 -> bf16) into LDS, coalesced.
  {
    const float* xsrc = xin + ((size_t)e * CAP_ + rowStart) * D_;
    #pragma unroll
    for (int i = 0; i < (BR * D_ / 2) / THREADS; ++i) {   // 64 pairs/thread
      int p  = tid + i * THREADS;
      int r  = p >> 8;              // / (D_/2)
      int d0 = (p & 255) * 2;
      float2 v = *(const float2*)(xsrc + (size_t)r * D_ + d0);
      *(unsigned*)&Xs[r * XP + d0] = pack2(v.x, v.y);
    }
  }
  __syncthreads();

  // ---- Load this wave's X A-fragments into registers, once.
  // Row m = tr*16 + l16 (all lanes), 16 K-steps of 32 -> 128 VGPRs.
  v16bf xfrag[D_ / 32];
  {
    const __bf16* arow = &Xs[(tr * 16 + l16) * XP];
    #pragma unroll
    for (int ks = 0; ks < D_ / 32; ++ks)
      xfrag[ks] = frag2(arow + ks * 32 + 8 * kg,
                        arow + ks * 32 + 16 + 8 * kg);
  }

  v8f acc[4][4] = {};              // persistent f32 Y accumulators

  const float* w1e = w1g + (size_t)e * D_ * H_;
  const float* w2e = w2g + (size_t)e * H_ * D_;

  for (int hc = 0; hc < H_; hc += HB) {
    __syncthreads();  // prior readers of W1sT/Xs alias, Ts, W2sT are done

    // Stage W1 chunk transposed: W1sT[h][d] = w1[e][d][hc+h]
    #pragma unroll
    for (int i = 0; i < (D_ * HB / 2) / THREADS; ++i) {   // 32 pairs/thread
      int p  = tid + i * THREADS;
      int h  = p & (HB - 1);          // consecutive tid -> consecutive h
      int d0 = (p >> 6) * 2;
      float a = w1e[(size_t)d0       * H_ + hc + h];
      float b = w1e[(size_t)(d0 + 1) * H_ + hc + h];
      *(unsigned*)&W1sT[h * W1P + d0] = pack2(a, b);
    }
    // Stage W2 chunk transposed: W2sT[n][k] = w2[e][hc+k][n]
    #pragma unroll
    for (int i = 0; i < (HB * D_ / 2) / THREADS; ++i) {   // 32 pairs/thread
      int p  = tid + i * THREADS;
      int n  = p & (D_ - 1);          // consecutive tid -> consecutive n
      int k0 = (p >> 9) * 2;
      float a = w2e[(size_t)(hc + k0)     * D_ + n];
      float b = w2e[(size_t)(hc + k0 + 1) * D_ + n];
      *(unsigned*)&W2sT[n * W2P + k0] = pack2(a, b);
    }

    // Prefetch next chunk's weight slices into cache (global_prefetch_b8);
    // fire-and-forget, hides the staging latency of the next barrier round.
    {
      int hcn = hc + HB;
      if (hcn < H_) {
        #pragma unroll
        for (int i = 0; i < 2; ++i) {          // W1: 512 rows x 2 lines
          int L = tid + i * THREADS;
          __builtin_prefetch(w1e + (size_t)(L >> 1) * H_ + hcn + (L & 1) * 32, 0, 1);
        }
        const float* w2n = w2e + (size_t)hcn * D_;
        #pragma unroll
        for (int i = 0; i < 2; ++i)            // W2: 128KB contiguous
          __builtin_prefetch(w2n + (size_t)(tid + i * THREADS) * 32, 0, 1);
      }
    }
    __syncthreads();

    // ---- GEMM1: two 16x16 T tiles per wave, K = D_ = 512.
    // A comes from registers (xfrag); only B is read from LDS.
    {
      v8f t0 = {}, t1 = {};
      const __bf16* brow0 = &W1sT[(tc0 * 16 + l16) * W1P];
      const __bf16* brow1 = brow0 + 16 * W1P;
      #pragma unroll
      for (int ks = 0; ks < D_ / 32; ++ks) {
        int ko = ks * 32 + 8 * kg;
        v16bf b0 = frag2(brow0 + ko, brow0 + ko + 16);
        v16bf b1 = frag2(brow1 + ko, brow1 + ko + 16);
        t0 = __builtin_amdgcn_wmma_f32_16x16x32_bf16(false, xfrag[ks], false, b0,
                                                     (short)0, t0, false, false);
        t1 = __builtin_amdgcn_wmma_f32_16x16x32_bf16(false, xfrag[ks], false, b1,
                                                     (short)0, t1, false, false);
      }
      // bias1 + exact GELU, write bf16 tiles to Ts
      float bias10 = b1g[(size_t)e * H_ + hc + tc0 * 16 + l16];
      float bias11 = b1g[(size_t)e * H_ + hc + tc0 * 16 + 16 + l16];
      #pragma unroll
      for (int r = 0; r < 8; ++r) {
        int m = tr * 16 + kg * 8 + r;
        Ts[m * TSP + tc0 * 16 + l16]      = (__bf16)gelu_exact(t0[r] + bias10);
        Ts[m * TSP + tc0 * 16 + 16 + l16] = (__bf16)gelu_exact(t1[r] + bias11);
      }
    }
    __syncthreads();

    // ---- GEMM2 accumulate: Y(64x64 per wave) += T(64x64) * W2chunk(64x64)
    #pragma unroll
    for (int kk = 0; kk < HB; kk += 32) {
      v16bf afr[4], bfr[4];
      #pragma unroll
      for (int i = 0; i < 4; ++i) {
        const __bf16* row = &Ts[(yr0 + i * 16 + l16) * TSP];
        afr[i] = frag2(row + kk + 8 * kg, row + kk + 16 + 8 * kg);
      }
      #pragma unroll
      for (int j = 0; j < 4; ++j) {
        const __bf16* row = &W2sT[(yc0 + j * 16 + l16) * W2P];
        bfr[j] = frag2(row + kk + 8 * kg, row + kk + 16 + 8 * kg);
      }
      #pragma unroll
      for (int i = 0; i < 4; ++i)
        #pragma unroll
        for (int j = 0; j < 4; ++j)
          acc[i][j] = __builtin_amdgcn_wmma_f32_16x16x32_bf16(
              false, afr[i], false, bfr[j], (short)0, acc[i][j], false, false);
    }
  }

  // ---- Epilogue: + b2, row mask, fp32 store
  float bias2[4];
  #pragma unroll
  for (int j = 0; j < 4; ++j)
    bias2[j] = b2g[(size_t)e * D_ + yc0 + j * 16 + l16];

  #pragma unroll
  for (int i = 0; i < 4; ++i) {
    #pragma unroll
    for (int r = 0; r < 8; ++r) {
      int m    = yr0 + i * 16 + kg * 8 + r;
      bool ok  = (rowStart + m) < valid;
      #pragma unroll
      for (int j = 0; j < 4; ++j) {
        float v = ok ? (acc[i][j][r] + bias2[j]) : 0.0f;
        outBlk[(size_t)m * D_ + yc0 + j * 16 + l16] = v;
      }
    }
  }
}

extern "C" void kernel_launch(void* const* d_in, const int* in_sizes, int n_in,
                              void* d_out, int out_size, void* d_ws, size_t ws_size,
                              hipStream_t stream) {
  (void)in_sizes; (void)n_in; (void)out_size; (void)d_ws; (void)ws_size;
  const float*     x  = (const float*)d_in[0];
  const long long* vl = (const long long*)d_in[1];   // jnp.int64
  const float*     w1 = (const float*)d_in[2];
  const float*     b1 = (const float*)d_in[3];
  const float*     w2 = (const float*)d_in[4];
  const float*     b2 = (const float*)d_in[5];
  float* out = (float*)d_out;

  dim3 grid(CAP_ / BR, E_);   // 8 row-blocks x 64 experts
  GroupFFNExperts_kernel<<<grid, dim3(THREADS), 0, stream>>>(x, vl, w1, b1, w2, b2, out);
}
